// SpaceAttention_1769526526211
// MI455X (gfx1250) — compile-verified, matrix operand order checked by reading, not run on hardware
//
#include <hip/hip_runtime.h>
#include <hip/hip_bf16.h>

typedef __attribute__((ext_vector_type(16))) _Float16 v16h;
typedef __attribute__((ext_vector_type(8)))  _Float16 v8h;
typedef __attribute__((ext_vector_type(4)))  _Float16 v4h;
typedef __attribute__((ext_vector_type(8)))  float    v8f;

#define E_DIM   1024
#define HEADS   16
#define HD      64
#define SEQ     256
#define MTOT    16384   /* 64 batches * 256 tokens */

// ---------------------------------------------------------------------------
// WMMA helpers. CDNA5 16x16x32 f16 fragment layout (ISA 7.12.2):
//  A (16x32, MxK): lane L -> row m = L&15; half = L>>4; the 16 f16 elements are
//  the row's K-chunks [half*8 .. half*8+7] and [16+half*8 .. 16+half*8+7]
//  -> two contiguous 16-byte loads. B (32x16, KxN) is symmetric with N=lane&15,
//  and since B[k][n] = W[n][k] for our "x @ W^T" GEMMs, B fragments are also
//  two contiguous 16-byte loads from W's row n.
// ---------------------------------------------------------------------------
__device__ __forceinline__ v16h load_frag(const _Float16* __restrict__ row, int half) {
  union { v16h v; v8h h[2]; } u;
  u.h[0] = *(const v8h*)(row + half * 8);
  u.h[1] = *(const v8h*)(row + 16 + half * 8);
  return u.v;
}

__device__ __forceinline__ v8f wmma_f16(v16h a, v16h b, v8f c) {
  return __builtin_amdgcn_wmma_f32_16x16x32_f16(false, a, false, b, (short)0, c,
                                                false, false);
}

// 64x64 output tile per wave: 4 A-frags + 4 B-frags -> 16 WMMAs per K-step.
// Load:WMMA ratio 1:2 (vs 2:1 with a single 16x16 tile).
__device__ __forceinline__ void gemm_64x64(const _Float16* __restrict__ A0,
                                           const _Float16* __restrict__ B0,
                                           int half, v8f acc[4][4]) {
  const size_t rstride = (size_t)16 * E_DIM;
#pragma unroll 2
  for (int k0 = 0; k0 < E_DIM; k0 += 32) {
    __builtin_prefetch(A0 + k0 + 512, 0, 1);  // global_prefetch_b8
    __builtin_prefetch(B0 + k0 + 512, 0, 1);
    v16h a[4], b[4];
#pragma unroll
    for (int t = 0; t < 4; ++t) {
      a[t] = load_frag(A0 + t * rstride + k0, half);
      b[t] = load_frag(B0 + t * rstride + k0, half);
    }
#pragma unroll
    for (int mt = 0; mt < 4; ++mt)
#pragma unroll
      for (int nt = 0; nt < 4; ++nt)
        acc[mt][nt] = wmma_f16(a[mt], b[nt], acc[mt][nt]);
  }
}

// ---------------------------------------------------------------------------
// Kernel 0: f32 -> f16 cast (x and the four weight matrices). n % 4 == 0.
// ---------------------------------------------------------------------------
__global__ __launch_bounds__(256) void cast_f32_to_f16(
    const float* __restrict__ src, _Float16* __restrict__ dst, int n) {
  int i = (blockIdx.x * 256 + threadIdx.x) * 4;
  if (i < n) {
    float4 f = *(const float4*)(src + i);
    v4h h = {(_Float16)f.x, (_Float16)f.y, (_Float16)f.z, (_Float16)f.w};
    *(v4h*)(dst + i) = h;
  }
}

// ---------------------------------------------------------------------------
// Kernel 1: fused QKV projection.
//  grid = (MTOT/64, HEADS/4, 3), block = 128 (4 wave32).
//  Each block computes 64 rows x 4 heads; wave w owns head h0+w (64 dims =
//  a 64x64 register tile, 4x4 WMMA accumulators).
//  z==0/1 (Q/K): RoPE + LayerNorm epilogue via LDS, all 128 threads.
//  z==2   (V)  : stores V transposed per head (dim-major).
// ---------------------------------------------------------------------------
__global__ __launch_bounds__(128) void qkv_gemm_kernel(
    const _Float16* __restrict__ X,
    const _Float16* __restrict__ Wq, const _Float16* __restrict__ Wk,
    const _Float16* __restrict__ Wv,
    const float* __restrict__ lnqg, const float* __restrict__ lnqb,
    const float* __restrict__ lnkg, const float* __restrict__ lnkb,
    _Float16* __restrict__ Qo, _Float16* __restrict__ Ko,
    _Float16* __restrict__ Vt) {
  __shared__ __attribute__((aligned(16))) float tile[4][64][65];
  const int wave = threadIdx.x >> 5;
  const int lane = threadIdx.x & 31;
  const int half = lane >> 4;
  const int idx  = lane & 15;
  const int m0   = blockIdx.x << 6;     // 64 rows per block
  const int h0   = blockIdx.y << 2;     // 4 heads per block
  const int z    = blockIdx.z;
  const int head = h0 + wave;

  const _Float16* W = (z == 0) ? Wq : ((z == 1) ? Wk : Wv);
  const _Float16* A0 = X + (size_t)(m0 + idx) * E_DIM;
  const _Float16* B0 = W + (size_t)(head * HD + idx) * E_DIM;

  v8f acc[4][4] = {};
  gemm_64x64(A0, B0, half, acc);

  if (z == 2) {  // V: store transposed (per-head dim-major): Vt[b][h][d][pos]
#pragma unroll
    for (int mt = 0; mt < 4; ++mt)
#pragma unroll
      for (int nt = 0; nt < 4; ++nt)
#pragma unroll
        for (int r = 0; r < 8; ++r) {
          int m = m0 + mt * 16 + r + 8 * half;
          int pos = m & (SEQ - 1);
          int batch = m >> 8;
          int dim = nt * 16 + idx;
          Vt[(((size_t)batch * HEADS + head) * HD + dim) * SEQ + pos] =
              (_Float16)acc[mt][nt][r];
        }
    return;
  }

  // Q/K: dump 64x64 wave tile to LDS, then RoPE + LN per row (64-dim head).
#pragma unroll
  for (int mt = 0; mt < 4; ++mt)
#pragma unroll
    for (int nt = 0; nt < 4; ++nt)
#pragma unroll
      for (int r = 0; r < 8; ++r)
        tile[wave][mt * 16 + r + 8 * half][nt * 16 + idx] = acc[mt][nt][r];
  __syncthreads();

  const float* g  = (z == 0) ? lnqg : lnkg;
  const float* bb = (z == 0) ? lnqb : lnkb;
  _Float16* base = (z == 0) ? Qo : Ko;
  for (int rid = threadIdx.x; rid < 256; rid += 128) {
    const int w  = rid >> 6;         // which wave tile (head h0+w)
    const int lr = rid & 63;         // local row
    const int m = m0 + lr;
    const int pos = m & (SEQ - 1);
    float rv[64];
    float mu = 0.f;
#pragma unroll 8
    for (int j = 0; j < 32; ++j) {   // rotate_half RoPE: pairs (j, j+32)
      float ang = (float)pos * __powf(10000.f, -(float)j * (1.0f / 32.0f));
      float c = __cosf(ang), s = __sinf(ang);
      float v1 = tile[w][lr][j], v2 = tile[w][lr][j + 32];
      rv[j]      = v1 * c - v2 * s;
      rv[j + 32] = v2 * c + v1 * s;
      mu += rv[j] + rv[j + 32];
    }
    mu *= (1.0f / 64.0f);
    float var = 0.f;
#pragma unroll
    for (int i = 0; i < 64; ++i) { float d = rv[i] - mu; var += d * d; }
    var *= (1.0f / 64.0f);
    float rs = rsqrtf(var + 1e-5f);
    _Float16* dst = base + (size_t)m * E_DIM + (h0 + w) * HD;
#pragma unroll
    for (int i = 0; i < 64; ++i)
      dst[i] = (_Float16)((rv[i] - mu) * rs * g[i] + bb[i]);
  }
}

// ---------------------------------------------------------------------------
// Kernel 2: attention per (query-tile, head, batch).
//  grid = (SEQ/16, HEADS, 64), block = 128.
//  Phase 1: S = Q Kt / 8 via WMMA (A-frags loop-invariant) -> LDS f32
//  Phase 2: parallel softmax (8 threads/row, LDS reductions) -> f16 P in LDS
//  Phase 3: O = P V via WMMA; A from LDS (ds_load_b128), B from transposed V.
// ---------------------------------------------------------------------------
__global__ __launch_bounds__(128) void attn_kernel(
    const _Float16* __restrict__ Q, const _Float16* __restrict__ K,
    const _Float16* __restrict__ Vt, _Float16* __restrict__ O) {
  __shared__ __attribute__((aligned(16))) float    sS[16][264];
  __shared__ __attribute__((aligned(16))) _Float16 sP[16][256];
  __shared__ float redm[16][8];
  __shared__ float reds[16][8];
  const int wave = threadIdx.x >> 5;
  const int lane = threadIdx.x & 31;
  const int half = lane >> 4;
  const int idx  = lane & 15;
  const int m0    = blockIdx.x << 4;
  const int head  = blockIdx.y;
  const int batch = blockIdx.z;

  const _Float16* arow =
      Q + ((size_t)(batch * SEQ + m0 + idx)) * E_DIM + head * HD;

  // Phase 1: scores (each wave does 4 of 16 key-tiles)
#pragma unroll
  for (int t = 0; t < 4; ++t) {
    int kbase = (wave + 4 * t) << 4;
    const _Float16* brow =
        K + ((size_t)(batch * SEQ + kbase + idx)) * E_DIM + head * HD;
    v8f acc = {};
    acc = wmma_f16(load_frag(arow, half),      load_frag(brow, half),      acc);
    acc = wmma_f16(load_frag(arow + 32, half), load_frag(brow + 32, half), acc);
#pragma unroll
    for (int r = 0; r < 8; ++r)
      sS[r + 8 * half][kbase + idx] = acc[r] * 0.125f;  // 1/sqrt(64)
  }
  __syncthreads();

  // Phase 2: softmax, 8 threads per row (each owns a 32-column segment)
  {
    const int row = threadIdx.x >> 3;
    const int seg = threadIdx.x & 7;
    const int c0 = seg * 32;
    float mx = -3.4e38f;
#pragma unroll 8
    for (int j = 0; j < 32; ++j) mx = fmaxf(mx, sS[row][c0 + j]);
    redm[row][seg] = mx;
    __syncthreads();
    float rmx = redm[row][0];
#pragma unroll
    for (int s = 1; s < 8; ++s) rmx = fmaxf(rmx, redm[row][s]);
    float sum = 0.f;
#pragma unroll 8
    for (int j = 0; j < 32; ++j) {
      float e = __expf(sS[row][c0 + j] - rmx);
      sS[row][c0 + j] = e;
      sum += e;
    }
    reds[row][seg] = sum;
    __syncthreads();
    float tot = 0.f;
#pragma unroll
    for (int s = 0; s < 8; ++s) tot += reds[row][s];
    float inv = 1.0f / tot;
#pragma unroll 8
    for (int j = 0; j < 32; ++j)
      sP[row][c0 + j] = (_Float16)(sS[row][c0 + j] * inv);
  }
  __syncthreads();

  // Phase 3: O = P @ V  (K-dim = 256 keys), A from LDS, B from transposed V.
  {
    const int nd0 = wave * 16;  // dim sub-tile within head
    const _Float16* vrow =
        Vt + (((size_t)batch * HEADS + head) * HD + nd0 + idx) * SEQ;
    v8f acc = {};
#pragma unroll
    for (int k0 = 0; k0 < SEQ; k0 += 32) {
      v16h a = load_frag((const _Float16*)&sP[idx][0] + k0, half);  // ds_load_b128
      v16h b = load_frag(vrow + k0, half);
      acc = wmma_f16(a, b, acc);
    }
#pragma unroll
    for (int r = 0; r < 8; ++r) {
      int m = m0 + r + 8 * half;
      O[((size_t)(batch * SEQ + m)) * E_DIM + head * HD + nd0 + idx] =
          (_Float16)acc[r];
    }
  }
}

// ---------------------------------------------------------------------------
// Kernel 3: output projection  out = O @ Wo^T  (f32 result to d_out).
//  grid = (MTOT/64, E/256), block = 128; 64x64 register tile per wave.
// ---------------------------------------------------------------------------
__global__ __launch_bounds__(128) void out_gemm_kernel(
    const _Float16* __restrict__ O, const _Float16* __restrict__ Wo,
    float* __restrict__ out) {
  const int wave = threadIdx.x >> 5;
  const int lane = threadIdx.x & 31;
  const int half = lane >> 4;
  const int idx  = lane & 15;
  const int m0 = blockIdx.x << 6;
  const int n0 = (blockIdx.y << 8) + wave * 64;
  const _Float16* A0 = O + (size_t)(m0 + idx) * E_DIM;
  const _Float16* B0 = Wo + (size_t)(n0 + idx) * E_DIM;
  v8f acc[4][4] = {};
  gemm_64x64(A0, B0, half, acc);
#pragma unroll
  for (int mt = 0; mt < 4; ++mt)
#pragma unroll
    for (int nt = 0; nt < 4; ++nt)
#pragma unroll
      for (int r = 0; r < 8; ++r)
        out[(size_t)(m0 + mt * 16 + r + 8 * half) * E_DIM + n0 + nt * 16 + idx] =
            acc[mt][nt][r];
}

// ---------------------------------------------------------------------------
extern "C" void kernel_launch(void* const* d_in, const int* in_sizes, int n_in,
                              void* d_out, int out_size, void* d_ws,
                              size_t ws_size, hipStream_t stream) {
  (void)in_sizes; (void)n_in; (void)out_size; (void)ws_size;
  const float* x    = (const float*)d_in[0];
  const float* Wq   = (const float*)d_in[1];
  const float* Wk   = (const float*)d_in[2];
  const float* Wv   = (const float*)d_in[3];
  const float* Wo   = (const float*)d_in[4];
  const float* lnqg = (const float*)d_in[5];
  const float* lnqb = (const float*)d_in[6];
  const float* lnkg = (const float*)d_in[7];
  const float* lnkb = (const float*)d_in[8];
  float* out = (float*)d_out;

  const size_t NE = (size_t)MTOT * E_DIM;    // 16,777,216 elements
  const size_t WE = (size_t)E_DIM * E_DIM;   //  1,048,576 elements

  _Float16* p    = (_Float16*)d_ws;
  _Float16* x16  = p;  p += NE;
  _Float16* wq16 = p;  p += WE;
  _Float16* wk16 = p;  p += WE;
  _Float16* wv16 = p;  p += WE;
  _Float16* wo16 = p;  p += WE;
  _Float16* q16  = p;  p += NE;
  _Float16* k16  = p;  p += NE;
  _Float16* vt16 = p;  p += NE;
  _Float16* o16  = p;  p += NE;

  // Stage 0: precision conversion (f32 -> f16) for activations and weights.
  cast_f32_to_f16<<<(int)(NE / 1024), 256, 0, stream>>>(x, x16, (int)NE);
  cast_f32_to_f16<<<(int)(WE / 1024), 256, 0, stream>>>(Wq, wq16, (int)WE);
  cast_f32_to_f16<<<(int)(WE / 1024), 256, 0, stream>>>(Wk, wk16, (int)WE);
  cast_f32_to_f16<<<(int)(WE / 1024), 256, 0, stream>>>(Wv, wv16, (int)WE);
  cast_f32_to_f16<<<(int)(WE / 1024), 256, 0, stream>>>(Wo, wo16, (int)WE);

  // Stage 1: fused QKV projection + RoPE + LayerNorm (Q,K) / transpose (V).
  qkv_gemm_kernel<<<dim3(MTOT / 64, HEADS / 4, 3), 128, 0, stream>>>(
      x16, wq16, wk16, wv16, lnqg, lnqb, lnkg, lnkb, q16, k16, vt16);

  // Stage 2: attention (scores -> softmax -> PV) per (qtile, head, batch).
  attn_kernel<<<dim3(SEQ / 16, HEADS, MTOT / SEQ), 128, 0, stream>>>(
      q16, k16, vt16, o16);

  // Stage 3: output projection to f32.
  out_gemm_kernel<<<dim3(MTOT / 64, E_DIM / 256), 128, 0, stream>>>(
      o16, wo16, out);
}
